// ResGatedConv_v3_17540646437070
// MI455X (gfx1250) — compile-verified, matrix operand order checked by reading, not run on hardware
//
#include <hip/hip_runtime.h>
#include <hip/hip_bf16.h>
#include <math.h>

// ---------------------------------------------------------------------------
// Problem constants (match reference)
// ---------------------------------------------------------------------------
#define NN    100000      // nodes
#define NE    1600000     // edges
#define NG    256         // graphs
#define C_IN  16
#define C_HID 64
#define EPSF  1e-5f

typedef __bf16 bf16_t;
typedef bf16_t v16bf __attribute__((ext_vector_type(16)));
typedef float  v8f   __attribute__((ext_vector_type(8)));

union BVec { uint4 u[2]; v16bf v; };

// ---------------------------------------------------------------------------
// Utility kernels
// ---------------------------------------------------------------------------
__global__ void k_zero(float* __restrict__ p, int n) {
    int i = blockIdx.x * blockDim.x + threadIdx.x;
    if (i < n) p[i] = 0.0f;
}

__global__ void k_counts(const int* __restrict__ batch, float* __restrict__ counts) {
    int i = blockIdx.x * blockDim.x + threadIdx.x;
    if (i < NN) atomicAdd(&counts[batch[i]], 1.0f);
}

// Zero-pad x [NN,16] -> xpad [NN,32] so layer-1 GEMM needs no K guards.
__global__ void k_pad_x(const float* __restrict__ x, float* __restrict__ xpad) {
    int i = blockIdx.x * blockDim.x + threadIdx.x;
    if (i >= NN * 32) return;
    int node = i >> 5, c = i & 31;
    xpad[i] = (c < C_IN) ? x[node * C_IN + c] : 0.0f;
}

// ---------------------------------------------------------------------------
// Pack weight W[Krows,N] (row-major f32) into per-lane WMMA B-operand layout:
// for (ntile, kstep, lane): 16 bf16 (32 bytes), element e covers
//   kk = kstep*32 + (lane>>4)*8 + e + (e>=8 ? 8 : 0),  col = ntile*16 + (lane&15)
// Rows kk >= Krows are zero-filled (handles K=16 padding).
// ---------------------------------------------------------------------------
__global__ void k_pack_b(const float* __restrict__ W, bf16_t* __restrict__ P,
                         int Krows, int Kpad, int N) {
    int t = blockIdx.x * blockDim.x + threadIdx.x;
    int KS = Kpad / 32;
    int total = (N / 16) * KS * 32;
    if (t >= total) return;
    int lane = t & 31;
    int ks   = (t >> 5) % KS;
    int nt   = (t >> 5) / KS;
    int half = lane >> 4, l16 = lane & 15;
    int col  = nt * 16 + l16;
    bf16_t* dst = P + (size_t)t * 16;
    for (int e = 0; e < 16; ++e) {
        int kk = ks * 32 + half * 8 + e + ((e >= 8) ? 8 : 0);
        float v = (kk < Krows) ? W[kk * N + col] : 0.0f;
        dst[e] = (bf16_t)v;
    }
}

// ---------------------------------------------------------------------------
// WMMA GEMM, no guards: Out[M,N] = act(A[M,K] @ B + bias).
// Requires M%16==0, N%16==0, K%32==0 (callers guarantee). One wave per
// 16x16 tile; K fully unrolled -> K/32 v_wmma_f32_16x16x32_bf16 per kernel.
// A loads: 4x float4 (unconditional, contiguous). B loads: 2x uint4 from
// the pre-packed operand buffer.
// ---------------------------------------------------------------------------
template <int K>
__global__ void k_wmma_gemm(const float* __restrict__ A,
                            const bf16_t* __restrict__ Bpack,
                            const float* __restrict__ bias,
                            float* __restrict__ Out,
                            int N, int act /*0=none,1=relu*/) {
    constexpr int KS = K / 32;
    const int lane = threadIdx.x;        // blockDim.x == 32
    const int half = lane >> 4;
    const int l16  = lane & 15;
    const int row0 = blockIdx.x * 16;
    const int nt   = blockIdx.y;
    const int col  = nt * 16 + l16;
    const float* arowp = A + (size_t)(row0 + l16) * K;

    v8f acc = {};
#pragma unroll
    for (int ks = 0; ks < KS; ++ks) {
        const float4* p0 = (const float4*)(arowp + ks * 32 + half * 8);
        const float4* p1 = (const float4*)(arowp + ks * 32 + 16 + half * 8);
        float4 f0 = p0[0], f1 = p0[1];
        float4 f2 = p1[0], f3 = p1[1];
        v16bf a;
        a[0]  = (bf16_t)f0.x; a[1]  = (bf16_t)f0.y; a[2]  = (bf16_t)f0.z; a[3]  = (bf16_t)f0.w;
        a[4]  = (bf16_t)f1.x; a[5]  = (bf16_t)f1.y; a[6]  = (bf16_t)f1.z; a[7]  = (bf16_t)f1.w;
        a[8]  = (bf16_t)f2.x; a[9]  = (bf16_t)f2.y; a[10] = (bf16_t)f2.z; a[11] = (bf16_t)f2.w;
        a[12] = (bf16_t)f3.x; a[13] = (bf16_t)f3.y; a[14] = (bf16_t)f3.z; a[15] = (bf16_t)f3.w;

        const uint4* bp = (const uint4*)(Bpack + (((size_t)nt * KS + ks) * 32 + lane) * 16);
        BVec b;
        b.u[0] = bp[0];
        b.u[1] = bp[1];

        acc = __builtin_amdgcn_wmma_f32_16x16x32_bf16(
            /*neg_a=*/false, a, /*neg_b=*/false, b.v,
            /*c_mod=*/(short)0, acc, /*reuse_a=*/false, /*reuse_b=*/false);
    }

    float bcol = bias ? bias[col] : 0.0f;
#pragma unroll
    for (int r = 0; r < 8; ++r) {
        float v = acc[r] + bcol;
        if (act) v = fmaxf(v, 0.0f);
        Out[(size_t)(row0 + half * 8 + r) * N + col] = v;
    }
}

// Final 64->3 layer: trivial scalar kernel (256x3 outputs).
__global__ void k_last(const float* __restrict__ A, const float* __restrict__ W,
                       const float* __restrict__ b, float* __restrict__ out) {
    int i = blockIdx.x * blockDim.x + threadIdx.x;
    if (i >= NG * 3) return;
    int r = i / 3, c = i % 3;
    float s = b[c];
    for (int k = 0; k < 64; ++k) s += A[r * 64 + k] * W[k * 3 + c];
    out[i] = s;
}

// ---------------------------------------------------------------------------
// Edge message + segment-sum (the bandwidth-bound phase).
// One wave per edge, lane handles 2 consecutive channels (float2 loads),
// f32 atomic scatter into agg[dst].
// ---------------------------------------------------------------------------
__global__ void k_edge(const int* __restrict__ ei,
                       const float* __restrict__ Kf,
                       const float* __restrict__ Qf,
                       const float* __restrict__ Vf,
                       float* __restrict__ agg) {
    int gid = blockIdx.x * blockDim.x + threadIdx.x;
    int e   = gid >> 5;              // 32 lanes per edge
    int c2  = (gid & 31) * 2;        // channel pair (C_HID = 64)
    if (e >= NE) return;
    int src = ei[e];
    int dst = ei[NE + e];
    float2 kv = *(const float2*)(Kf + dst * C_HID + c2);
    float2 qv = *(const float2*)(Qf + src * C_HID + c2);
    float2 vv = *(const float2*)(Vf + src * C_HID + c2);
    float g0 = 1.0f / (1.0f + __expf(-(kv.x + qv.x)));
    float g1 = 1.0f / (1.0f + __expf(-(kv.y + qv.y)));
    atomicAdd(agg + dst * C_HID + c2,     g0 * vv.x);
    atomicAdd(agg + dst * C_HID + c2 + 1, g1 * vv.y);
}

// ---------------------------------------------------------------------------
// GraphNorm pipeline (3 element-wise passes + 2 tiny per-graph kernels)
// ---------------------------------------------------------------------------
__global__ void k_gelu_skip_sum(float* __restrict__ agg,   // in: agg, out: h
                                const float* __restrict__ S,
                                const float* __restrict__ cbias,
                                const int* __restrict__ batch,
                                float* __restrict__ gsum) {
    int i = blockIdx.x * blockDim.x + threadIdx.x;
    if (i >= NN * C_HID) return;
    int node = i >> 6, c = i & 63;
    float h = agg[i] + S[i] + cbias[c];
    h = 0.5f * h * (1.0f + erff(h * 0.70710678118654752f));  // exact GELU
    agg[i] = h;
    atomicAdd(&gsum[batch[node] * C_HID + c], h);
}

__global__ void k_div_counts(float* __restrict__ g, const float* __restrict__ counts) {
    int i = blockIdx.x * blockDim.x + threadIdx.x;
    if (i >= NG * C_HID) return;
    float cnt = fmaxf(counts[i >> 6], 1.0f);
    g[i] = g[i] / cnt;
}

__global__ void k_center_sqsum(const float* __restrict__ h,
                               const float* __restrict__ gmean,
                               const float* __restrict__ mscale,
                               const int* __restrict__ batch,
                               float* __restrict__ o,
                               float* __restrict__ gsq) {
    int i = blockIdx.x * blockDim.x + threadIdx.x;
    if (i >= NN * C_HID) return;
    int node = i >> 6, c = i & 63;
    int g = batch[node];
    float v = h[i] - gmean[g * C_HID + c] * mscale[c];
    o[i] = v;
    atomicAdd(&gsq[g * C_HID + c], v * v);
}

__global__ void k_var_to_inv(float* __restrict__ gsq, const float* __restrict__ counts) {
    int i = blockIdx.x * blockDim.x + threadIdx.x;
    if (i >= NG * C_HID) return;
    float cnt = fmaxf(counts[i >> 6], 1.0f);
    gsq[i] = rsqrtf(gsq[i] / cnt + EPSF);
}

__global__ void k_gnorm_apply(const float* __restrict__ o,
                              const float* __restrict__ ginv,
                              const float* __restrict__ w,
                              const float* __restrict__ b,
                              const int* __restrict__ batch,
                              float* __restrict__ xout) {
    int i = blockIdx.x * blockDim.x + threadIdx.x;
    if (i >= NN * C_HID) return;
    int node = i >> 6, c = i & 63;
    int g = batch[node];
    xout[i] = w[c] * o[i] * ginv[g * C_HID + c] + b[c];
}

__global__ void k_pool(const float* __restrict__ x, const int* __restrict__ batch,
                       float* __restrict__ pooled) {
    int i = blockIdx.x * blockDim.x + threadIdx.x;
    if (i >= NN * C_HID) return;
    int node = i >> 6, c = i & 63;
    atomicAdd(&pooled[batch[node] * C_HID + c], x[i]);
}

// LayerNorm over feature dim; one thread per row (rows=256, trivial).
__global__ void k_layernorm(float* __restrict__ X, const float* __restrict__ gamma,
                            const float* __restrict__ beta, int rows, int F) {
    int r = blockIdx.x * blockDim.x + threadIdx.x;
    if (r >= rows) return;
    float mu = 0.0f;
    for (int f = 0; f < F; ++f) mu += X[r * F + f];
    mu /= (float)F;
    float var = 0.0f;
    for (int f = 0; f < F; ++f) { float d = X[r * F + f] - mu; var += d * d; }
    var /= (float)F;
    float inv = rsqrtf(var + EPSF);
    for (int f = 0; f < F; ++f)
        X[r * F + f] = gamma[f] * (X[r * F + f] - mu) * inv + beta[f];
}

// ---------------------------------------------------------------------------
// Host-side orchestration
// ---------------------------------------------------------------------------
static inline int cdiv(int a, int b) { return (a + b - 1) / b; }

extern "C" void kernel_launch(void* const* d_in, const int* in_sizes, int n_in,
                              void* d_out, int out_size, void* d_ws, size_t ws_size,
                              hipStream_t stream) {
    (void)in_sizes; (void)n_in; (void)out_size; (void)ws_size;

    // --- input pointer map (top-level: setup_inputs insertion order;
    //     params: JAX pytree flatten = sorted dict keys at each level) ---
    const float* x     = (const float*)d_in[0];
    const int*   ei    = (const int*)d_in[1];
    const int*   batch = (const int*)d_in[2];

    // conv[l]: bias, key.W, key.b, query.W, query.b, skip.W, value.W, value.b
    const float *convBias[3], *Wk[3], *bk[3], *Wq[3], *bq[3], *Wsk[3], *Wv[3], *bv[3];
    for (int l = 0; l < 3; ++l) {
        int b = 3 + l * 8;
        convBias[l] = (const float*)d_in[b + 0];
        Wk[l]  = (const float*)d_in[b + 1];  bk[l] = (const float*)d_in[b + 2];
        Wq[l]  = (const float*)d_in[b + 3];  bq[l] = (const float*)d_in[b + 4];
        Wsk[l] = (const float*)d_in[b + 5];
        Wv[l]  = (const float*)d_in[b + 6];  bv[l] = (const float*)d_in[b + 7];
    }
    // gn[l]: bias, mean_scale, weight
    const float *gnBias[3], *gnMS[3], *gnW[3];
    for (int l = 0; l < 3; ++l) {
        int b = 27 + l * 3;
        gnBias[l] = (const float*)d_in[b + 0];
        gnMS[l]   = (const float*)d_in[b + 1];
        gnW[l]    = (const float*)d_in[b + 2];
    }
    const float* h0W   = (const float*)d_in[36];
    const float* h0b   = (const float*)d_in[37];
    const float* h1W   = (const float*)d_in[38];
    const float* h1b   = (const float*)d_in[39];
    const float* lastW = (const float*)d_in[40];
    const float* lastb = (const float*)d_in[41];
    const float* ln0be = (const float*)d_in[42];
    const float* ln0ga = (const float*)d_in[43];
    const float* ln1be = (const float*)d_in[44];
    const float* ln1ga = (const float*)d_in[45];

    // --- workspace carve-up ---
    float* ws = (float*)d_ws;
    size_t off = 0;
    const size_t NC = (size_t)NN * C_HID;
    float* xcur   = ws + off; off += NC;
    float* Kbuf   = ws + off; off += NC;
    float* Qbuf   = ws + off; off += NC;
    float* Vbuf   = ws + off; off += NC;
    float* Sbuf   = ws + off; off += NC;   // skip proj, later reused for centered o
    float* agg    = ws + off; off += NC;   // agg, later h
    float* xpad   = ws + off; off += (size_t)NN * 32;
    float* gsum   = ws + off; off += (size_t)NG * C_HID;
    float* gsq    = ws + off; off += (size_t)NG * C_HID;
    float* counts = ws + off; off += NG;
    float* pooled = ws + off; off += (size_t)NG * C_HID;
    float* mlp0   = ws + off; off += (size_t)NG * 128;
    float* mlp1   = ws + off; off += (size_t)NG * 64;
    off = (off + 15) & ~(size_t)15;        // align packed-weight region (64B)
    bf16_t* pk = (bf16_t*)(ws + off);
    size_t poff = 0;
    bf16_t *pKQVS[3][4], *pH0, *pH1;
    const int convKpad[3] = {32, 64, 64};
    for (int l = 0; l < 3; ++l)
        for (int m = 0; m < 4; ++m) { pKQVS[l][m] = pk + poff; poff += (size_t)convKpad[l] * C_HID; }
    pH0 = pk + poff; poff += 64 * 128;
    pH1 = pk + poff; poff += 128 * 64;

    const int TB  = 256;
    const int nNC = NN * C_HID;            // 6.4M
    const int nGC = NG * C_HID;

    // --- one-shot preparation (cheap; part of every deterministic launch) ---
    k_pad_x<<<cdiv(NN * 32, TB), TB, 0, stream>>>(x, xpad);
    const float* convW[3][4];
    for (int l = 0; l < 3; ++l) {
        convW[l][0] = Wk[l]; convW[l][1] = Wq[l]; convW[l][2] = Wv[l]; convW[l][3] = Wsk[l];
        int krows = (l == 0) ? C_IN : C_HID;
        int kpad  = convKpad[l];
        int tot   = (C_HID / 16) * (kpad / 32) * 32;
        for (int m = 0; m < 4; ++m)
            k_pack_b<<<cdiv(tot, TB), TB, 0, stream>>>(convW[l][m], pKQVS[l][m], krows, kpad, C_HID);
    }
    k_pack_b<<<cdiv((128 / 16) * 2 * 32, TB), TB, 0, stream>>>(h0W, pH0, 64, 64, 128);
    k_pack_b<<<cdiv((64 / 16) * 4 * 32, TB), TB, 0, stream>>>(h1W, pH1, 128, 128, 64);

    k_zero<<<cdiv(NG, TB), TB, 0, stream>>>(counts, NG);
    k_counts<<<cdiv(NN, TB), TB, 0, stream>>>(batch, counts);

    // --- conv layers ---
    for (int l = 0; l < 3; ++l) {
        dim3 gg(NN / 16, C_HID / 16);      // 6250 x 4 tiles
        if (l == 0) {
            k_wmma_gemm<32><<<gg, 32, 0, stream>>>(xpad, pKQVS[0][0], bk[0], Kbuf, C_HID, 0);
            k_wmma_gemm<32><<<gg, 32, 0, stream>>>(xpad, pKQVS[0][1], bq[0], Qbuf, C_HID, 0);
            k_wmma_gemm<32><<<gg, 32, 0, stream>>>(xpad, pKQVS[0][2], bv[0], Vbuf, C_HID, 0);
            k_wmma_gemm<32><<<gg, 32, 0, stream>>>(xpad, pKQVS[0][3], nullptr, Sbuf, C_HID, 0);
        } else {
            k_wmma_gemm<64><<<gg, 32, 0, stream>>>(xcur, pKQVS[l][0], bk[l], Kbuf, C_HID, 0);
            k_wmma_gemm<64><<<gg, 32, 0, stream>>>(xcur, pKQVS[l][1], bq[l], Qbuf, C_HID, 0);
            k_wmma_gemm<64><<<gg, 32, 0, stream>>>(xcur, pKQVS[l][2], bv[l], Vbuf, C_HID, 0);
            k_wmma_gemm<64><<<gg, 32, 0, stream>>>(xcur, pKQVS[l][3], nullptr, Sbuf, C_HID, 0);
        }

        k_zero<<<cdiv(nNC, TB), TB, 0, stream>>>(agg, nNC);
        k_edge<<<cdiv(NE * 32, TB), TB, 0, stream>>>(ei, Kbuf, Qbuf, Vbuf, agg);

        // GraphNorm
        k_zero<<<cdiv(nGC, TB), TB, 0, stream>>>(gsum, nGC);
        k_gelu_skip_sum<<<cdiv(nNC, TB), TB, 0, stream>>>(agg, Sbuf, convBias[l], batch, gsum);
        k_div_counts<<<cdiv(nGC, TB), TB, 0, stream>>>(gsum, counts);
        k_zero<<<cdiv(nGC, TB), TB, 0, stream>>>(gsq, nGC);
        k_center_sqsum<<<cdiv(nNC, TB), TB, 0, stream>>>(agg, gsum, gnMS[l], batch, Sbuf, gsq);
        k_var_to_inv<<<cdiv(nGC, TB), TB, 0, stream>>>(gsq, counts);
        k_gnorm_apply<<<cdiv(nNC, TB), TB, 0, stream>>>(Sbuf, gsq, gnW[l], gnBias[l], batch, xcur);
    }

    // --- global mean pool ---
    k_zero<<<cdiv(nGC, TB), TB, 0, stream>>>(pooled, nGC);
    k_pool<<<cdiv(nNC, TB), TB, 0, stream>>>(xcur, batch, pooled);
    k_div_counts<<<cdiv(nGC, TB), TB, 0, stream>>>(pooled, counts);

    // --- MLP head (256 rows = 16 WMMA M-tiles) ---
    {
        dim3 g0(NG / 16, 128 / 16);
        k_wmma_gemm<64><<<g0, 32, 0, stream>>>(pooled, pH0, h0b, mlp0, 128, 1);
        k_layernorm<<<1, NG, 0, stream>>>(mlp0, ln0ga, ln0be, NG, 128);

        dim3 g1(NG / 16, 64 / 16);
        k_wmma_gemm<128><<<g1, 32, 0, stream>>>(mlp0, pH1, h1b, mlp1, 64, 1);
        k_layernorm<<<1, NG, 0, stream>>>(mlp1, ln1ga, ln1be, NG, 64);

        k_last<<<cdiv(NG * 3, TB), TB, 0, stream>>>(mlp1, lastW, lastb, (float*)d_out);
    }
}
